// ARNN2_5196910428289
// MI455X (gfx1250) — matrix operand order repaired and checked by reading, not compiled, not run on hardware
//
#include <hip/hip_runtime.h>
#include <hip/hip_bf16.h>
#include <math.h>

// ---- problem sizes (fixed by the reference) ----
#define BB 256
#define TT 128
#define II 128
#define HH 256
#define OO 128
#define NPONDER 8
#define THRESH 0.99f   // 1 - EPS

typedef __attribute__((ext_vector_type(16))) __bf16 v16bf;
typedef __attribute__((ext_vector_type(8)))  __bf16 v8bf;
typedef __attribute__((ext_vector_type(4)))  __bf16 v4bf;
typedef __attribute__((ext_vector_type(8)))  float  v8f;

#define L2E 1.4426950408889634f

// ---- fast transcendentals: single TRANS32 ops on CDNA5 ----
__device__ __forceinline__ float fast_exp2(float x) { return __builtin_amdgcn_exp2f(x); }
__device__ __forceinline__ float fast_rcp(float x)  { return __builtin_amdgcn_rcpf(x); }

__device__ __forceinline__ float fast_tanh(float x) {
#if __has_builtin(__builtin_amdgcn_tanhf)
    return __builtin_amdgcn_tanhf(x);            // v_tanh_f32
#elif __has_builtin(__builtin_amdgcn_tanh_f32)
    return __builtin_amdgcn_tanh_f32(x);
#else
    float t = fast_exp2(2.0f * L2E * x);
    return 1.0f - 2.0f * fast_rcp(t + 1.0f);
#endif
}

__device__ __forceinline__ float fast_sigmoid(float x) {
    return fast_rcp(1.0f + fast_exp2(-L2E * x));
}

__device__ __forceinline__ v8f wmma_bf16(v16bf a, v16bf b, v8f c) {
    return __builtin_amdgcn_wmma_f32_16x16x32_bf16(
        false, a, false, b, (short)0, c, false, false);
}

// A-matrix fragment (16x32 bf16, MxK): row = lane%16, 16 contiguous K per lane half.
__device__ __forceinline__ v16bf load_a_frag(const __bf16* src, int row, int ld,
                                             int kf, int hi) {
    union { v16bf v; v8bf h[2]; } u;
    const __bf16* p = src + row * ld + kf * 32 + hi * 16;
    u.h[0] = *(const v8bf*)(p);
    u.h[1] = *(const v8bf*)(p + 8);
    return u.v;
}

// B-matrix fragment (32x16 bf16, KxN): lane = column N, 16 contiguous K per lane
// (lanes 0-15: K=k0..k0+15, lanes 16-31: K=k0+16..k0+31) — from bf16 LDS copy.
__device__ __forceinline__ v16bf load_b_frag_lds(const __bf16* W, int n, int ldk, int k0) {
    union { v16bf v; v8bf h[2]; } u;
    const __bf16* p = W + n * ldk + k0;
    u.h[0] = *(const v8bf*)(p);
    u.h[1] = *(const v8bf*)(p + 8);
    return u.v;
}

// Same fragment layout, loaded once from global f32 weights.
__device__ __forceinline__ v16bf load_b_frag_global(const float* __restrict__ W,
                                                    int n, int ldk, int k0) {
    union { v16bf v; __bf16 e[16]; } u;
    const float* p = W + (size_t)n * ldk + k0;
#pragma unroll
    for (int i = 0; i < 16; ++i) u.e[i] = (__bf16)p[i];
    return u.v;
}

__global__ __launch_bounds__(512, 1)
void act_rnn_persistent(const float* __restrict__ x,   const float* __restrict__ s0,
                        const float* __restrict__ Wih, const float* __restrict__ bih,
                        const float* __restrict__ Whh, const float* __restrict__ bhh,
                        const float* __restrict__ Wo,  const float* __restrict__ bo,
                        const float* __restrict__ Wh,  const float* __restrict__ bh,
                        float* __restrict__ out, float* __restrict__ pond_out) {
    // dynamic LDS: [ Wo bf16 128x256 | Wih(cols 1..128) bf16 256x128 ]  (128KB)
    extern __shared__ __align__(16) __bf16 dyn_lds_base[];
    __bf16* wo_lds  = dyn_lds_base;                 // [no][k], ld = HH
    __bf16* wih_lds = dyn_lds_base + OO * HH;       // [n][k],  ld = II

    __shared__ __attribute__((aligned(16))) __bf16 h_lds[2][16 * HH];   // double-buffered state
    __shared__ __attribute__((aligned(16))) __bf16 xt_lds[2][16 * II];  // double-buffered x_t
    __shared__ __attribute__((aligned(16))) __bf16 whbf[HH];            // halt weights (bf16)
    __shared__ float bias_lds[HH];   // bih + bhh
    __shared__ float wcol0[HH];      // Wih[:,0] (ponder-flag column)
    __shared__ float bo_lds[OO];
    __shared__ float p_arr[16];
    __shared__ float halted[16], haltedp[16], haltsum[16], accum[16], pondc[16], warr[16];
    __shared__ float bh_s;

    const int tid  = threadIdx.x;
    const int lane = tid & 31;
    const int wave = tid >> 5;          // 16 waves
    const int row  = lane & 15;
    const int hi   = lane >> 4;
    const int b0   = blockIdx.x * 16;
    const int n    = wave * 16 + row;         // cell column 0..255
    const int no   = (wave & 7) * 16 + row;   // output column 0..127

    // ---- one-time init ----
    for (int i = tid; i < HH; i += 512) {
        bias_lds[i] = bih[i] + bhh[i];
        wcol0[i]    = Wih[(size_t)i * (II + 1)];
        whbf[i]     = (__bf16)Wh[i];
    }
    for (int i = tid; i < OO; i += 512) bo_lds[i] = bo[i];
    for (int i = tid; i < OO * HH; i += 512) wo_lds[i] = (__bf16)Wo[i];   // row-major [no][k]
    for (int i = tid; i < HH * II; i += 512)                              // skip flag column
        wih_lds[i] = (__bf16)Wih[(size_t)(i >> 7) * (II + 1) + 1 + (i & 127)];
    if (tid == 0) bh_s = bh[0];
    for (int i = tid; i < 16 * HH; i += 512)
        h_lds[0][i] = (__bf16)s0[(size_t)(b0 + (i >> 8)) * HH + (i & 255)];
    {   // stage x_0: thread -> row=wave, cols lane*4..lane*4+3
        const float4 v = *(const float4*)(x + (size_t)(b0 + wave) * TT * II + lane * 4);
        v4bf c; c[0]=(__bf16)v.x; c[1]=(__bf16)v.y; c[2]=(__bf16)v.z; c[3]=(__bf16)v.w;
        *(v4bf*)(&xt_lds[0][wave * II + lane * 4]) = c;
    }

    // ---- resident weight fragments: only the hot Whh strip lives in VGPRs ----
    v16bf whhF[8];   // Whh^T : K=256, this wave's 16-col strip
#pragma unroll
    for (int kf = 0; kf < 8; ++kf) whhF[kf] = load_b_frag_global(Whh, n, HH, kf * 32 + hi * 16);
    __syncthreads();

    // loop-invariant scalars hoisted to registers
    const float bias_n = bias_lds[n];
    const float w0_n   = wcol0[n];
    const float bos    = bo_lds[no];
    const float bhv    = bh_s;
    union { v8bf v; __bf16 e[8]; } wha, whb;    // halt-weight chunks (waves 8-15 path)
    wha.v = *(const v8bf*)(whbf + (lane & 15) * 16);
    whb.v = *(const v8bf*)(whbf + (lane & 15) * 16 + 8);

    for (int t = 0; t < TT; ++t) {
        const int cur = t & 1;

        // prefetch x_{t+1} into registers; committed to LDS at end of timestep
        float4 xpre = {0.f, 0.f, 0.f, 0.f};
        if (t + 1 < TT)
            xpre = *(const float4*)(x + (size_t)(b0 + wave) * TT * II
                                      + (size_t)(t + 1) * II + lane * 4);

        // hoisted input GEMM: xw = x_t @ Wih[:,1:]^T  (two chains, B from LDS)
        v8f xw;
        {
            const __bf16* xb = xt_lds[cur];
            v8f c0 = {}, c1 = {};
            c0 = wmma_bf16(load_a_frag(xb, row, II, 0, hi),
                           load_b_frag_lds(wih_lds, n, II, 0 * 32 + hi * 16), c0);
            c1 = wmma_bf16(load_a_frag(xb, row, II, 1, hi),
                           load_b_frag_lds(wih_lds, n, II, 1 * 32 + hi * 16), c1);
            c0 = wmma_bf16(load_a_frag(xb, row, II, 2, hi),
                           load_b_frag_lds(wih_lds, n, II, 2 * 32 + hi * 16), c0);
            c1 = wmma_bf16(load_a_frag(xb, row, II, 3, hi),
                           load_b_frag_lds(wih_lds, n, II, 3 * 32 + hi * 16), c1);
            xw = c0;
#pragma unroll
            for (int e = 0; e < 8; ++e) xw[e] += c1[e];
        }

        v8f sacc = {}, oacc = {};
        int curh = 0;
        for (int p = 0; p < NPONDER; ++p) {
            const __bf16* hr  = h_lds[curh];
            __bf16*       hwr = h_lds[curh ^ 1];

            // ---- RNN cell: two-chain WMMA, chain 0 seeded with xw ----
            v8f c0 = xw, c1 = {};
#pragma unroll
            for (int kf = 0; kf < 8; kf += 2) {
                c0 = wmma_bf16(load_a_frag(hr, row, HH, kf,     hi), whhF[kf],     c0);
                c1 = wmma_bf16(load_a_frag(hr, row, HH, kf + 1, hi), whhF[kf + 1], c1);
            }
            const float bn = bias_n + ((p == 0) ? w0_n : 0.0f);
            float snm[8];
#pragma unroll
            for (int e = 0; e < 8; ++e) {
                const int M = e + hi * 8;
                const float m = (p == 0) ? 1.0f : (1.0f - halted[M]);
                snm[e] = fast_tanh(c0[e] + c1[e] + bn) * m;     // masked s_n
            }
#pragma unroll
            for (int e = 0; e < 8; ++e)
                hwr[(e + hi * 8) * HH + n] = (__bf16)snm[e];    // write ALT buffer (no pre-barrier)
            __syncthreads();   // B1: new h visible

            // ---- concurrently: out GEMM (waves 0-7) + halt dot (waves 8-15) ----
            v8f o0 = {}, o1 = {};
            if (wave < 8) {
#pragma unroll
                for (int kf = 0; kf < 8; kf += 2) {
                    o0 = wmma_bf16(load_a_frag(hwr, row, HH, kf,     hi),
                                   load_b_frag_lds(wo_lds, no, HH, kf * 32 + hi * 16),       o0);
                    o1 = wmma_bf16(load_a_frag(hwr, row, HH, kf + 1, hi),
                                   load_b_frag_lds(wo_lds, no, HH, (kf + 1) * 32 + hi * 16), o1);
                }
            } else {
                const int g = (wave - 8) * 2 + hi;   // batch row 0..15
                const int j = lane & 15;
                union { v8bf v; __bf16 e[8]; } ha, hb;
                ha.v = *(const v8bf*)(hwr + g * HH + j * 16);
                hb.v = *(const v8bf*)(hwr + g * HH + j * 16 + 8);
                float part = 0.0f;
#pragma unroll
                for (int c = 0; c < 8; ++c)
                    part += (float)ha.e[c] * (float)wha.e[c] + (float)hb.e[c] * (float)whb.e[c];
                part += __shfl_xor(part, 8, 32);
                part += __shfl_xor(part, 4, 32);
                part += __shfl_xor(part, 2, 32);
                part += __shfl_xor(part, 1, 32);
                if (j == 0) p_arr[g] = fast_sigmoid(part + bhv);
            }
            __syncthreads();   // B2: p_arr ready

            // ---- halting automaton (one thread per batch row) ----
            if (tid < 16) {
                const float pn = p_arr[tid];
                if (p == 0) {
                    haltedp[tid] = 0.0f;
                    haltsum[tid] = pn;
                    accum[tid]   = pn;
                    halted[tid]  = (pn >= THRESH) ? 1.0f : 0.0f;
                    const float wb2 = (pn > THRESH) ? 1.0f : 0.0f;
                    pondc[tid]   = wb2;
                    warr[tid]    = wb2;
                } else {
                    const float hin = halted[tid];
                    haltedp[tid] = hin;
                    const float tmp = ((1.0f - hin) * (haltsum[tid] + pn) >= THRESH) ? 1.0f : 0.0f;
                    haltsum[tid] += pn;
                    accum[tid]   += pn * (1.0f - hin);
                    halted[tid]   = hin + tmp;
                    const float wb2 = (accum[tid] > THRESH) ? 1.0f : 0.0f;
                    pondc[tid]   += 1.0f - wb2;
                    warr[tid]     = wb2;
                }
            }
            __syncthreads();   // B3: halting state ready

            // ---- accumulate halting-weighted state / output ----
#pragma unroll
            for (int e = 0; e < 8; ++e) {
                const int M = e + hi * 8;
                const float w = warr[M];
                sacc[e] += snm[e] * w;
                if (wave < 8)
                    oacc[e] += w * (o0[e] + o1[e] + bos * (1.0f - haltedp[M]));
            }
            curh ^= 1;
        }

        // ---- emit outputs; carry s_acc (buffer 0); commit x prefetch ----
        if (wave < 8) {
#pragma unroll
            for (int e = 0; e < 8; ++e) {
                const int M = e + hi * 8;
                out[(size_t)(b0 + M) * TT * OO + (size_t)t * OO + no] = oacc[e];
            }
        }
        if (tid < 16) pond_out[(size_t)(b0 + tid) * TT + t] = pondc[tid];
#pragma unroll
        for (int e = 0; e < 8; ++e)
            h_lds[0][(e + hi * 8) * HH + n] = (__bf16)sacc[e];
        if (t + 1 < TT) {
            v4bf c; c[0]=(__bf16)xpre.x; c[1]=(__bf16)xpre.y; c[2]=(__bf16)xpre.z; c[3]=(__bf16)xpre.w;
            *(v4bf*)(&xt_lds[1 - cur][wave * II + lane * 4]) = c;
        }
        __syncthreads();
    }
}

extern "C" void kernel_launch(void* const* d_in, const int* in_sizes, int n_in,
                              void* d_out, int out_size, void* d_ws, size_t ws_size,
                              hipStream_t stream) {
    const float* x   = (const float*)d_in[0];
    const float* s0  = (const float*)d_in[1];
    const float* Wih = (const float*)d_in[2];
    const float* bih = (const float*)d_in[3];
    const float* Whh = (const float*)d_in[4];
    const float* bhh = (const float*)d_in[5];
    const float* Wo  = (const float*)d_in[6];
    const float* bo  = (const float*)d_in[7];
    const float* Wh  = (const float*)d_in[8];
    const float* bh  = (const float*)d_in[9];
    float* out  = (float*)d_out;
    float* pond = out + (size_t)BB * TT * OO;

    // 64KB Wo + 64KB Wih (bf16) in dynamic LDS
    const size_t dyn_lds = ((size_t)OO * HH + (size_t)HH * II) * sizeof(__hip_bfloat16);

    hipLaunchKernelGGL(act_rnn_persistent, dim3(BB / 16), dim3(512), dyn_lds, stream,
                       x, s0, Wih, bih, Whh, bhh, Wo, bo, Wh, bh, out, pond);
}
